// ParametrizedSpinGlassHamiltonian_53429393162649
// MI455X (gfx1250) — matrix operand order repaired and checked by reading, not compiled, not run on hardware
//
#include <hip/hip_runtime.h>
#include <hip/hip_bf16.h>

typedef __attribute__((ext_vector_type(16))) _Float16 v16h;
typedef __attribute__((ext_vector_type(8)))  float    v8f;
typedef int v4i __attribute__((vector_size(4 * sizeof(int))));   // matches builtin param type

#define NSPIN   64
#define TILE_M  16
#define LDSS    68      // padded LDS row stride in floats (16B-aligned rows, conflict-free)
#define WPB     4       // waves per block (128 threads)

#ifndef __has_builtin
#define __has_builtin(x) 0
#endif
#if __has_builtin(__builtin_amdgcn_global_load_async_to_lds_b128)
#define USE_ASYNC_LDS 1
#else
#define USE_ASYNC_LDS 0
#endif

// ---------------------------------------------------------------------------
// Prep: scatter 2016 lower-tri couplings into a symmetric f16 64x64 J matrix.
// tril_indices(k=-1) row-major flat index of (i,j), i>j:  i*(i-1)/2 + j
// ---------------------------------------------------------------------------
__global__ void build_J_kernel(const float* __restrict__ couplings,
                               _Float16* __restrict__ Jh) {
    int t = blockIdx.x * blockDim.x + threadIdx.x;
    if (t >= NSPIN * NSPIN) return;
    int i = t >> 6, j = t & 63;
    float v = 0.0f;
    if (i > j)      v = couplings[(i * (i - 1)) / 2 + j];
    else if (j > i) v = couplings[(j * (j - 1)) / 2 + i];
    Jh[t] = (_Float16)v;
}

__device__ __forceinline__ void lds_fence() {
    // Compiler-only fence: LDS ops from one wave execute in order (ISA 7.3),
    // so cross-lane store->load within a wave only needs compile-order.
    asm volatile("" ::: "memory");
}

__device__ __forceinline__ void wait_async_le8() {
#if USE_ASYNC_LDS
    asm volatile("s_wait_asynccnt 0x8" ::: "memory");
#endif
}
__device__ __forceinline__ void wait_async_le0() {
#if USE_ASYNC_LDS
    asm volatile("s_wait_asynccnt 0x0" ::: "memory");
#endif
}

// Issue one 16x64 f32 states tile into an LDS buffer.
// Async path: GLOBAL_LOAD_ASYNC_TO_LDS_B128 (ASYNCcnt, no VGPR staging).
// Fallback:  global_load_b128 -> ds_store_b128.
__device__ __forceinline__ void issue_tile(const float* __restrict__ states,
                                           int tile, float* dst, int lane) {
    const float4* g = (const float4*)(states + (size_t)tile * (TILE_M * NSPIN));
#pragma unroll
    for (int i = 0; i < 8; ++i) {
        int f   = i * 32 + lane;           // 0..255 float4s, coalesced
        int row = f >> 4, c4 = f & 15;
        float* lp = dst + row * LDSS + c4 * 4;
#if USE_ASYNC_LDS
        __builtin_amdgcn_global_load_async_to_lds_b128(
            (__attribute__((address_space(1))) v4i*)(g + f),
            (__attribute__((address_space(3))) v4i*)lp,
            0, 0);
#else
        *(float4*)lp = g[f];
#endif
    }
}

__device__ __forceinline__ void cvt16(v16h& a, float4 r0, float4 r1,
                                      float4 r2, float4 r3) {
    a[0]=(_Float16)r0.x;  a[1]=(_Float16)r0.y;  a[2] =(_Float16)r0.z;  a[3] =(_Float16)r0.w;
    a[4]=(_Float16)r1.x;  a[5]=(_Float16)r1.y;  a[6] =(_Float16)r1.z;  a[7] =(_Float16)r1.w;
    a[8]=(_Float16)r2.x;  a[9]=(_Float16)r2.y;  a[10]=(_Float16)r2.z;  a[11]=(_Float16)r2.w;
    a[12]=(_Float16)r3.x; a[13]=(_Float16)r3.y; a[14]=(_Float16)r3.z;  a[15]=(_Float16)r3.w;
}

// ---------------------------------------------------------------------------
// Main kernel: one wave processes 16 states per iteration, double-buffered.
//   P' = S(16x64,f16) @ J(64x64,f16) + fields broadcast into accumulator
//   E  = -(rowwise sum_n S[m,n]*P'[m,n])   (f32 finish via LDS)
// ---------------------------------------------------------------------------
__global__ void __launch_bounds__(128)
ising_wmma_kernel(const float* __restrict__ states,
                  const float* __restrict__ fields,
                  const _Float16* __restrict__ Jh,
                  float* __restrict__ out, int ntiles) {
    __shared__ __align__(16) float sS[2][WPB][TILE_M * LDSS];  // states (double-buffered)
    __shared__ __align__(16) float sP[WPB][TILE_M * LDSS];     // P' tile (f32)

    const int lane = threadIdx.x & 31;
    const int w    = threadIdx.x >> 5;
    const int hi   = lane >> 4;      // lane half: 0 or 1
    const int ln   = lane & 15;

    // ---- B fragments: J columns, resident in registers the whole kernel ----
    // B 32x16 f16, k order used here (A same): lanes 0-15 col n=ln hold K=k0..k0+15 (2/VGPR),
    // lanes 16-31 hold K=k0+16..k0+31.  J symmetric -> column n == row n,
    // so each lane reads 16 contiguous f16 (32B, 32B-aligned).
    v16h bf[4][2];
#pragma unroll
    for (int t = 0; t < 4; ++t) {
#pragma unroll
        for (int kc = 0; kc < 2; ++kc) {
            int n  = t * 16 + ln;
            int k0 = kc * 32 + hi * 16;
            bf[t][kc] = *(const v16h*)(Jh + n * NSPIN + k0);
        }
    }
    // fields, one value per output column (C/D layout: n = ln per lane)
    float fld[4];
#pragma unroll
    for (int t = 0; t < 4; ++t) fld[t] = fields[t * 16 + ln];

    const int waveId = blockIdx.x * WPB + w;
    const int nWaves = gridDim.x * WPB;

    int tile = waveId;
    int buf  = 0;
    if (tile < ntiles) issue_tile(states, tile, &sS[0][w][0], lane);

    for (; tile < ntiles; tile += nWaves) {
        // ---- prefetch next tile into the alternate buffer, then wait for
        //      the current tile's 8 async loads (in-order completion) ----
        int nxt = tile + nWaves;
        if (nxt < ntiles) {
            issue_tile(states, nxt, &sS[buf ^ 1][w][0], lane);
            wait_async_le8();
        } else {
            wait_async_le0();
        }
        lds_fence();

        const float* sSw = &sS[buf][w][0];

        // ---- A fragments (16x32 f16). Layout: lane ln (low half)
        // holds row ln, K = cb..cb+15, cb = 16*hi (+32*kc).
        v16h a[2];
        const float* srow = sSw + ln * LDSS;
#pragma unroll
        for (int kc = 0; kc < 2; ++kc) {
            int cb = kc * 32 + hi * 16;
            float4 r0 = *(const float4*)(srow + cb);
            float4 r1 = *(const float4*)(srow + cb + 4);
            float4 r2 = *(const float4*)(srow + cb + 8);
            float4 r3 = *(const float4*)(srow + cb + 12);
            cvt16(a[kc], r0, r1, r2, r3);
        }

        // ---- WMMA: P'[16x64] in four 16x16 f32 tiles, fields folded in ----
#pragma unroll
        for (int t = 0; t < 4; ++t) {
            v8f c;
#pragma unroll
            for (int r = 0; r < 8; ++r) c[r] = fld[t];
            c = __builtin_amdgcn_wmma_f32_16x16x32_f16(false, a[0], false, bf[t][0],
                                                       (short)0, c, false, false);
            c = __builtin_amdgcn_wmma_f32_16x16x32_f16(false, a[1], false, bf[t][1],
                                                       (short)0, c, false, false);
            // C/D layout: VGPR r, low lanes -> (m=r, n=ln); high -> (m=r+8, n=ln)
#pragma unroll
            for (int r = 0; r < 8; ++r) {
                int m = r + hi * 8;
                sP[w][m * LDSS + t * 16 + ln] = c[r];
            }
        }
        lds_fence();

        // ---- E[m] = -(sum_n S[m,n]*P'[m,n]); lane ln handles row ln,
        //      half 'hi' covers columns [32*hi, 32*hi+32) ----
        const float* pp = &sP[w][ln * LDSS + hi * 32];
        const float* ss = sSw + ln * LDSS + hi * 32;
        float acc = 0.0f;
#pragma unroll
        for (int q = 0; q < 8; ++q) {
            float4 pv = *(const float4*)(pp + q * 4);
            float4 sv = *(const float4*)(ss + q * 4);
            acc = fmaf(pv.x, sv.x, acc);
            acc = fmaf(pv.y, sv.y, acc);
            acc = fmaf(pv.z, sv.z, acc);
            acc = fmaf(pv.w, sv.w, acc);
        }
        float tot = acc + __shfl_xor(acc, 16, 32);
        if (lane < 16) out[(size_t)tile * 16 + ln] = -tot;
        lds_fence();   // keep next iteration's LDS writes from crossing the reads
        buf ^= 1;
    }
}

// ---------------------------------------------------------------------------
// Scalar fallback for a non-multiple-of-16 tail (not hit for BATCH=1M).
// ---------------------------------------------------------------------------
__global__ void ising_tail_kernel(const float* __restrict__ states,
                                  const float* __restrict__ fields,
                                  const _Float16* __restrict__ Jh,
                                  float* __restrict__ out, int start, int batch) {
    int b = start + blockIdx.x * blockDim.x + threadIdx.x;
    if (b >= batch) return;
    const float* s = states + (size_t)b * NSPIN;
    float e = 0.0f, hf = 0.0f;
    for (int i = 0; i < NSPIN; ++i) {
        float si = s[i];
        hf = fmaf(si, fields[i], hf);
        float acc = 0.0f;
        for (int j = 0; j < NSPIN; ++j)
            acc = fmaf((float)Jh[i * NSPIN + j], s[j], acc);
        e = fmaf(si, acc, e);
    }
    out[b] = -e - hf;
}

extern "C" void kernel_launch(void* const* d_in, const int* in_sizes, int n_in,
                              void* d_out, int out_size, void* d_ws, size_t ws_size,
                              hipStream_t stream) {
    const float* states    = (const float*)d_in[0];
    const float* couplings = (const float*)d_in[1];
    const float* fields    = (const float*)d_in[2];
    float*       out       = (float*)d_out;
    _Float16*    Jh        = (_Float16*)d_ws;   // 64*64*2 = 8 KB

    const int batch  = in_sizes[0] / NSPIN;
    const int ntiles = batch / TILE_M;
    const int rem    = batch - ntiles * TILE_M;

    build_J_kernel<<<(NSPIN * NSPIN + 255) / 256, 256, 0, stream>>>(couplings, Jh);

    if (ntiles > 0) {
        int grid = (ntiles + WPB - 1) / WPB;
        if (grid > 8192) grid = 8192;           // persistent waves, B-frags amortized
        ising_wmma_kernel<<<grid, 32 * WPB, 0, stream>>>(states, fields, Jh, out, ntiles);
    }
    if (rem > 0) {
        ising_tail_kernel<<<1, 64, 0, stream>>>(states, fields, Jh, out,
                                                ntiles * TILE_M, batch);
    }
}
